// GPT2BBTAttention_55181739819321
// MI455X (gfx1250) — matrix operand order repaired and checked by reading, not compiled, not run on hardware
//
#include <hip/hip_runtime.h>
#include <hip/hip_bf16.h>

// MI455X / gfx1250 implementation: bf16 WMMA (f32 accum) everywhere.
// Pipeline: cvt->bf16, latent GEMM (scale folded), vT GEMM, fused
// scores+causal+softmax (streaming, per-wave q-block), att@v GEMM with
// causal K-limit + row-scale epilogue, final out GEMM to f32.
// K-loops are 2x-unrolled / register-double-buffered so global load latency
// is hidden behind each 8-WMMA batch (kEnd is always a multiple of 64 here).

typedef __attribute__((ext_vector_type(16))) __bf16 v16bf;
typedef __attribute__((ext_vector_type(8)))  __bf16 v8bf;
typedef __attribute__((ext_vector_type(8)))  float  v8f;

// ---- WMMA fragment loaders (layouts per CDNA5 ISA 7.12.2) -------------------
// A 16x32 bf16: lane l(0..15) row M=l, half h=lane>>4:
//   elems 0..7  = K kBase + h*8 + 0..7
//   elems 8..15 = K kBase + 16 + h*8 + 0..7
static __device__ __forceinline__ v16bf load_a_frag(const __bf16* rowPtr, int kBase, int h) {
  const __bf16* p = rowPtr + kBase + h * 8;
  v8bf lo = *(const v8bf*)p;
  v8bf hi = *(const v8bf*)(p + 16);
  v16bf r;
#pragma unroll
  for (int i = 0; i < 8; ++i) { r[i] = lo[i]; r[8 + i] = hi[i]; }
  return r;
}
// B 32x16 bf16 from row-major (N x K) weight: lane col n = l, half h:
//   elems 0..7 = K kBase + h*8 + 0..7, elems 8..15 = K kBase + 16 + h*8 + 0..7
static __device__ __forceinline__ v16bf load_b_frag(const __bf16* colPtr, int kBase, int h) {
  const __bf16* p = colPtr + kBase + h * 8;
  v8bf lo = *(const v8bf*)p;
  v8bf hi = *(const v8bf*)(p + 16);
  v16bf r;
#pragma unroll
  for (int i = 0; i < 8; ++i) { r[i] = lo[i]; r[8 + i] = hi[i]; }
  return r;
}

static __device__ __forceinline__ v8f wmma_bf16(v16bf a, v16bf b, v8f c) {
  return __builtin_amdgcn_wmma_f32_16x16x32_bf16(false, a, false, b, (short)0, c, false, false);
}

// ---- fp32 -> bf16 conversion ------------------------------------------------
__global__ void cvt_f32_bf16(const float* __restrict__ x, __bf16* __restrict__ y, long n) {
  long i = (long)blockIdx.x * blockDim.x + threadIdx.x;
  long stride = (long)gridDim.x * blockDim.x;
  for (; i < n; i += stride) y[i] = (__bf16)x[i];
}

// ---- Generic batched NT GEMM: D = scale * rowScale[m] * (A @ B^T) + bias ----
// A: (M x K) row-major (+ z*sA), B: (N x K) row-major (+ z*sB).
// WG tile 128x128, 8 waves (2x4), wave tile 64x32 (4x2 accumulators).
// All of M,N multiples of 16; K and causal kEnd multiples of 64.
__global__ void __launch_bounds__(256)
gemm_nt_bf16(const __bf16* __restrict__ A, const __bf16* __restrict__ B,
             const float* __restrict__ bias, const float* __restrict__ rowScale,
             float* __restrict__ Df, __bf16* __restrict__ Dbf,
             int M, int N, int K,
             long sA, long sB, long sD,
             int lda, int ldb, int ldd,
             float scale, int transD, int causal) {
  const int lane = threadIdx.x & 31;
  const int wave = threadIdx.x >> 5;
  const int wM = wave >> 2;   // 0..1
  const int wN = wave & 3;    // 0..3
  const int l = lane & 15;
  const int h = lane >> 4;
  const int z = blockIdx.z;
  const int mBase = blockIdx.y * 128;
  const int nBase = blockIdx.x * 128;

  const __bf16* Az = A + (long)z * sA;
  const __bf16* Bz = B + (long)z * sB;

  int row16[4], col16[2];
  bool bValid[2];
  const __bf16* Arow[4];
  const __bf16* Bcol[2];
#pragma unroll
  for (int mi = 0; mi < 4; ++mi) {
    row16[mi] = mBase + wM * 64 + mi * 16;
    Arow[mi] = Az + (long)(row16[mi] + l) * lda;
  }
#pragma unroll
  for (int ni = 0; ni < 2; ++ni) {
    col16[ni] = nBase + wN * 32 + ni * 16;
    bValid[ni] = (col16[ni] < N);   // uniform: N is a multiple of 16
    Bcol[ni] = Bz + (long)(col16[ni] + l) * ldb;
  }

  v8f acc[4][2];
#pragma unroll
  for (int mi = 0; mi < 4; ++mi)
#pragma unroll
    for (int ni = 0; ni < 2; ++ni)
#pragma unroll
      for (int r = 0; r < 8; ++r) acc[mi][ni][r] = 0.0f;

  const int kEnd = causal ? (K < mBase + 128 ? K : mBase + 128) : K;

  // Two register-resident fragment sets for software pipelining.
  v16bf a0[4], a1[4], b0[2], b1[2];
#pragma unroll
  for (int ni = 0; ni < 2; ++ni)
#pragma unroll
    for (int i = 0; i < 16; ++i) { b0[ni][i] = (__bf16)0.0f; b1[ni][i] = (__bf16)0.0f; }

  auto loadA = [&](v16bf (&a)[4], int k) {
#pragma unroll
    for (int mi = 0; mi < 4; ++mi) a[mi] = load_a_frag(Arow[mi], k, h);
  };
  auto loadB = [&](v16bf (&b)[2], int k) {
#pragma unroll
    for (int ni = 0; ni < 2; ++ni)
      if (bValid[ni]) b[ni] = load_b_frag(Bcol[ni], k, h);
  };
  auto mmastep = [&](v16bf (&a)[4], v16bf (&b)[2]) {
#pragma unroll
    for (int mi = 0; mi < 4; ++mi)
#pragma unroll
      for (int ni = 0; ni < 2; ++ni)
        acc[mi][ni] = wmma_bf16(a[mi], b[ni], acc[mi][ni]);
  };

  loadA(a0, 0);
  loadB(b0, 0);
  for (int k = 0; k < kEnd; k += 64) {
    // Issue loads for slab k+32 (always in range: kEnd % 64 == 0).
    loadA(a1, k + 32);
    loadB(b1, k + 32);
    if (k + 128 < kEnd) {   // gfx1250 global_prefetch two slabs ahead
      __builtin_prefetch(Arow[0] + k + 128, 0, 1);
      __builtin_prefetch(Bcol[0] + k + 128, 0, 1);
    }
    mmastep(a0, b0);        // consumes slab k while k+32 loads are in flight
    if (k + 64 < kEnd) {
      loadA(a0, k + 64);
      loadB(b0, k + 64);
    }
    mmastep(a1, b1);        // consumes slab k+32 while k+64 loads are in flight
  }

  // Epilogue. C/D layout: VGPR r, lane -> row = r + 8*h, col = l.
#pragma unroll
  for (int mi = 0; mi < 4; ++mi) {
#pragma unroll
    for (int ni = 0; ni < 2; ++ni) {
      if (!bValid[ni]) continue;
      const int col = col16[ni] + l;
      const float badd = bias ? bias[col] : 0.0f;
#pragma unroll
      for (int r = 0; r < 8; ++r) {
        const int row = row16[mi] + r + 8 * h;
        float v = acc[mi][ni][r];
        if (rowScale) v *= rowScale[(long)z * M + row];
        v = v * scale + badd;
        if (Df)          Df[(long)z * sD + (long)row * ldd + col] = v;
        else if (transD) Dbf[(long)z * sD + (long)col * ldd + row] = (__bf16)v;
        else             Dbf[(long)z * sD + (long)row * ldd + col] = (__bf16)v;
      }
    }
  }
}

// ---- Fused scores + causal mask + softmax -----------------------------------
// latent is pre-scaled by sqrt(1/8) (folded into the latent GEMM epilogue), so
// S = latent' @ latent'^T directly. Each wave owns one 16-row q-block; two
// streaming passes over key tiles (max, then exp/sum/store). Writes
// UNNORMALIZED probs (<=1) to att (bf16) and 1/rowsum to invSum; normalization
// is applied in the att@v GEMM epilogue.
__global__ void __launch_bounds__(256)
attn_scores_softmax(const __bf16* __restrict__ latent, __bf16* __restrict__ att,
                    float* __restrict__ invSum, int T) {
  const int R = 64;
  const int lane = threadIdx.x & 31;
  const int wave = threadIdx.x >> 5;
  const int l = lane & 15;
  const int h = lane >> 4;
  const int z = blockIdx.y;
  const int qb = blockIdx.x * 8 + wave;   // q tile index
  const int qBase = qb * 16;

  const __bf16* Lz = latent + (long)z * T * R;
  __bf16* Az = att + (long)z * T * T;
  float* Sz = invSum + (long)z * T;

  const __bf16* qp = Lz + (long)(qBase + l) * R;
  const v16bf a0 = load_a_frag(qp, 0, h);
  const v16bf a1 = load_a_frag(qp, 32, h);

  float rmax[8];
#pragma unroll
  for (int r = 0; r < 8; ++r) rmax[r] = -3.0e38f;

  // Pass 1: running row max over causal key range (double-buffered key tiles).
  {
    v16bf bc0 = load_b_frag(Lz + (long)l * R, 0, h);
    v16bf bc1 = load_b_frag(Lz + (long)l * R, 32, h);
    for (int j = 0; j <= qb; ++j) {
      v16bf bn0 = bc0, bn1 = bc1;
      if (j < qb) {
        const __bf16* kp = Lz + (long)((j + 1) * 16 + l) * R;
        bn0 = load_b_frag(kp, 0, h);
        bn1 = load_b_frag(kp, 32, h);
      }
      v8f s;
#pragma unroll
      for (int r = 0; r < 8; ++r) s[r] = 0.0f;
      s = wmma_bf16(a0, bc0, s);
      s = wmma_bf16(a1, bc1, s);
      if (j < qb) {
#pragma unroll
        for (int r = 0; r < 8; ++r) rmax[r] = fmaxf(rmax[r], s[r]);
      } else {
#pragma unroll
        for (int r = 0; r < 8; ++r)
          if (l <= r + 8 * h) rmax[r] = fmaxf(rmax[r], s[r]);  // col<=row on diagonal tile
      }
      bc0 = bn0; bc1 = bn1;
    }
  }
#pragma unroll
  for (int r = 0; r < 8; ++r)
    for (int m = 1; m < 16; m <<= 1)
      rmax[r] = fmaxf(rmax[r], __shfl_xor(rmax[r], m, 32));

  // Pass 2: recompute, exp, accumulate sums, store unnormalized probs.
  float rsum[8];
#pragma unroll
  for (int r = 0; r < 8; ++r) rsum[r] = 0.0f;
  {
    v16bf bc0 = load_b_frag(Lz + (long)l * R, 0, h);
    v16bf bc1 = load_b_frag(Lz + (long)l * R, 32, h);
    for (int j = 0; j <= qb; ++j) {
      v16bf bn0 = bc0, bn1 = bc1;
      if (j < qb) {
        const __bf16* kp = Lz + (long)((j + 1) * 16 + l) * R;
        bn0 = load_b_frag(kp, 0, h);
        bn1 = load_b_frag(kp, 32, h);
      }
      v8f s;
#pragma unroll
      for (int r = 0; r < 8; ++r) s[r] = 0.0f;
      s = wmma_bf16(a0, bc0, s);
      s = wmma_bf16(a1, bc1, s);
#pragma unroll
      for (int r = 0; r < 8; ++r) {
        const bool ok = (j < qb) || (l <= r + 8 * h);
        float p = ok ? __expf(s[r] - rmax[r]) : 0.0f;
        rsum[r] += p;
        Az[(long)(qBase + r + 8 * h) * T + j * 16 + l] = (__bf16)p;
      }
      bc0 = bn0; bc1 = bn1;
    }
  }
#pragma unroll
  for (int r = 0; r < 8; ++r)
    for (int m = 1; m < 16; m <<= 1)
      rsum[r] += __shfl_xor(rsum[r], m, 32);
  if (l == 0) {
#pragma unroll
    for (int r = 0; r < 8; ++r) Sz[qBase + r + 8 * h] = 1.0f / rsum[r];
  }
}

// ---- Host launcher ----------------------------------------------------------
extern "C" void kernel_launch(void* const* d_in, const int* in_sizes, int n_in,
                              void* d_out, int out_size, void* d_ws, size_t ws_size,
                              hipStream_t stream) {
  (void)in_sizes; (void)n_in; (void)out_size; (void)ws_size;
  const float* hs = (const float*)d_in[0];
  const float* Wb = (const float*)d_in[1];
  const float* Wv = (const float*)d_in[2];
  const float* bv = (const float*)d_in[3];
  const float* Wo = (const float*)d_in[4];
  const float* bo = (const float*)d_in[5];
  float* out = (float*)d_out;

  const int Bn = 4, T = 2048, C = 1024, R = 64;
  const long BT = (long)Bn * T;   // 8192

  char* ws = (char*)d_ws;
  size_t off = 0;
  auto alloc = [&](size_t bytes) -> char* {
    char* p = ws + off;
    off += (bytes + 255) & ~(size_t)255;
    return p;
  };
  __bf16* hs_b = (__bf16*)alloc((size_t)BT * C * 2);       // 16 MB
  __bf16* Wb_b = (__bf16*)alloc((size_t)R * C * 2);        // 128 KB
  __bf16* Wv_b = (__bf16*)alloc((size_t)C * C * 2);        // 2 MB
  __bf16* Wo_b = (__bf16*)alloc((size_t)C * C * 2);        // 2 MB
  __bf16* lat  = (__bf16*)alloc((size_t)BT * R * 2);       // 1 MB
  __bf16* vT   = (__bf16*)alloc((size_t)Bn * C * T * 2);   // 16 MB  (B,C,T)
  __bf16* att  = (__bf16*)alloc((size_t)Bn * T * T * 2);   // 32 MB
  float*  isum = (float*) alloc((size_t)Bn * T * 4);       // 32 KB
  __bf16* yb   = (__bf16*)alloc((size_t)BT * C * 2);       // 16 MB

  // 1) fp32 -> bf16 conversions
  cvt_f32_bf16<<<4096, 256, 0, stream>>>(hs, hs_b, BT * C);
  cvt_f32_bf16<<<256,  256, 0, stream>>>(Wb, Wb_b, (long)R * C);
  cvt_f32_bf16<<<1024, 256, 0, stream>>>(Wv, Wv_b, (long)C * C);
  cvt_f32_bf16<<<1024, 256, 0, stream>>>(Wo, Wo_b, (long)C * C);

  // 2) latent' = (X @ Wb^T) * sqrt(1/8)   -> (8192 x 64) bf16
  gemm_nt_bf16<<<dim3(1, (int)(BT / 128), 1), 256, 0, stream>>>(
      hs_b, Wb_b, nullptr, nullptr, nullptr, lat,
      (int)BT, R, C, 0, 0, 0, C, C, R, 0.35355339059327f, 0, 0);

  // 3) vT[b] = (X[b] @ Wv^T + bv)^T  -> (B, C, T) bf16 (transposed store)
  gemm_nt_bf16<<<dim3(C / 128, T / 128, Bn), 256, 0, stream>>>(
      hs_b, Wv_b, bv, nullptr, nullptr, vT,
      T, C, C, (long)T * C, 0, (long)C * T, C, C, T, 1.0f, 1, 0);

  // 4) att = softmax(causal(latent' @ latent'^T)) (unnormalized probs + 1/sum)
  hipMemsetAsync(att, 0, (size_t)Bn * T * T * 2, stream);
  attn_scores_softmax<<<dim3(T / 128, Bn), 256, 0, stream>>>(lat, att, isum, T);

  // 5) y[b] = diag(isum[b]) * (att[b] @ vT[b]^T)  -> (B, T, C) bf16, causal K
  gemm_nt_bf16<<<dim3(C / 128, T / 128, Bn), 256, 0, stream>>>(
      att, vT, nullptr, isum, nullptr, yb,
      T, C, T, (long)T * T, (long)C * T, (long)T * C, T, T, C, 1.0f, 0, 1);

  // 6) out = y @ Wo^T + bo  -> (8192 x 1024) f32
  gemm_nt_bf16<<<dim3(C / 128, (int)(BT / 128), 1), 256, 0, stream>>>(
      yb, Wo_b, bo, nullptr, out, nullptr,
      (int)BT, C, C, 0, 0, 0, C, C, C, 1.0f, 0, 0);
}
